// LocalAttentionEncoder_50388556317038
// MI455X (gfx1250) — matrix operand order repaired and checked by reading, not compiled, not run on hardware
//
#include <hip/hip_runtime.h>

// ---------------------------------------------------------------------------
// CDNA5 / gfx1250 local-attention encoder.
// GEMMs + attention contractions via v_wmma_f32_16x16x32_bf16.
// GEMM weight tiles staged with global_load_async_to_lds_b128 (ASYNCcnt).
// ---------------------------------------------------------------------------

typedef __bf16 bf16_t;
typedef __attribute__((ext_vector_type(16))) __bf16 v16bf;
typedef __attribute__((ext_vector_type(8)))  __bf16 v8bf;
typedef __attribute__((ext_vector_type(8)))  float  v8f;

#define WMMA_BF16(a, b, c) \
  __builtin_amdgcn_wmma_f32_16x16x32_bf16(false, (a), false, (b), (short)0, (c), false, false)

static __device__ __forceinline__ v16bf pack16(v8bf lo, v8bf hi) {
  v16bf r;
#pragma unroll
  for (int i = 0; i < 8; ++i) { r[i] = lo[i]; r[8 + i] = hi[i]; }
  return r;
}

static __device__ __forceinline__ int clampi(int x, int lo, int hi) {
  return x < lo ? lo : (x > hi ? hi : x);
}

// Async global->LDS copy of 16 bytes per lane (gfx1250 GLOBAL_LOAD_ASYNC_TO_LDS_B128).
static __device__ __forceinline__ void async_copy_b128(unsigned lds_addr, const void* gptr) {
  const unsigned long long ga = (unsigned long long)(size_t)gptr;
  asm volatile("global_load_async_to_lds_b128 %0, %1, off"
               :: "v"(lds_addr), "v"(ga) : "memory");
}
static __device__ __forceinline__ void wait_async0() {
  asm volatile("s_wait_asynccnt 0x0" ::: "memory");
}

// ---------------------------------------------------------------------------
// C = A (M x K, row-major bf16) * W^T (W is N x K row-major bf16) + bias
// Per block: 128 x 64 tile, 8 waves, each wave a 16 x 64 strip.
// 64-wide K step: W 64x64 tile staged via two async b128 copies per thread,
// then 2 k-steps x 4 n-tiles = 8 WMMAs per barrier pair.
// M,N multiples of 128/64; K multiple of 64.
// ---------------------------------------------------------------------------
__global__ __launch_bounds__(256, 1) void gemm_nt(
    const bf16_t* __restrict__ A, const bf16_t* __restrict__ W,
    const float* __restrict__ bias, float* __restrict__ Cf,
    bf16_t* __restrict__ Cb, int M, int N, int K, int relu)
{
  __shared__ bf16_t lB[64][80];  // [out-col][k 0..63], row stride 160B (16B aligned)
  const int tid = threadIdx.x;
  const int wv  = tid >> 5, lane = tid & 31;
  const int h16 = lane >> 4, l16 = lane & 15;
  const int m0  = blockIdx.x * 128 + wv * 16;
  const int n0  = blockIdx.y * 64;

  v8f acc[4];
#pragma unroll
  for (int t = 0; t < 4; ++t)
#pragma unroll
    for (int r = 0; r < 8; ++r) acc[t][r] = 0.f;

  const int br = tid >> 2;         // 0..63 : out-col within tile
  const int bc = (tid & 3) * 8;    // 0,8,16,24 : k within half-chunk
  const unsigned ldsB0 = (unsigned)(size_t)&lB[br][bc];       // LDS byte address
  const unsigned ldsB1 = (unsigned)(size_t)&lB[br][bc + 32];

  for (int kb = 0; kb < K; kb += 64) {
    __syncthreads();  // readers of previous tile done
    const bf16_t* wrow = W + (size_t)(n0 + br) * K + kb;
    async_copy_b128(ldsB0, wrow + bc);
    async_copy_b128(ldsB1, wrow + bc + 32);
    if (kb + 64 < K)
      __builtin_prefetch(wrow + 64 + bc, 0, 1);
    wait_async0();
    __syncthreads();  // tile visible to all waves

#pragma unroll
    for (int ks = 0; ks < 2; ++ks) {
      // A fragment: lane holds row m = l16, ks-local k = h16*8 + [0..7], +16..23
      const bf16_t* ap = A + (size_t)(m0 + l16) * K + kb + ks * 32 + h16 * 8;
      v16bf afr = pack16(*(const v8bf*)ap, *(const v8bf*)(ap + 16));

      // Preload all four B fragments, then issue the WMMAs back-to-back.
      v16bf bfr[4];
#pragma unroll
      for (int t = 0; t < 4; ++t) {
        const bf16_t* bp = &lB[t * 16 + l16][ks * 32 + h16 * 8];
        bfr[t] = pack16(*(const v8bf*)bp, *(const v8bf*)(bp + 16));
      }
#pragma unroll
      for (int t = 0; t < 4; ++t)
        acc[t] = WMMA_BF16(afr, bfr[t], acc[t]);
    }
  }

#pragma unroll
  for (int t = 0; t < 4; ++t) {
    const int col = n0 + t * 16 + l16;
    const float bval = bias[col];
#pragma unroll
    for (int r = 0; r < 8; ++r) {
      const int row = m0 + h16 * 8 + r;   // C/D layout: vgpr r -> M = r + 8*half
      float v = acc[t][r] + bval;
      if (relu) v = fmaxf(v, 0.f);
      const size_t o = (size_t)row * N + col;
      if (Cf) Cf[o] = v;
      if (Cb) Cb[o] = (bf16_t)v;
    }
  }
}

// ---------------------------------------------------------------------------
// Flash-style local windowed attention.
// grid = B*H*NW (=256) blocks of 256 threads. Wave wv owns 32 query rows.
// q,k,v stored bf16 in (b, n, h, d) layout (row stride 1024, head off h*64).
// V chunk (32 keys x 64 dims) is cooperatively transposed into LDS once per
// block so B-fragments of P*V become contiguous ds_load_b128 reads.
// ---------------------------------------------------------------------------
__global__ __launch_bounds__(256, 1) void local_attn(
    const bf16_t* __restrict__ Qm, const bf16_t* __restrict__ Km,
    const bf16_t* __restrict__ Vm, float* __restrict__ Om)
{
  __shared__ bf16_t lP[8][16][40];  // per-wave 16x32 softmax-prob tile (bf16)
  __shared__ bf16_t lV[64][40];     // V chunk transposed: [d][key]

  const int blk  = blockIdx.x;
  const int wwin = blk & 7;          // window (NW=8)
  const int hh   = (blk >> 3) & 15;  // head
  const int bb   = blk >> 7;         // batch
  const int tid  = threadIdx.x;
  const int wv   = tid >> 5, lane = tid & 31;
  const int h16  = lane >> 4, l16 = lane & 15;

  const size_t base = (size_t)bb * 2048 * 1024 + (size_t)hh * 64;
  const bf16_t* Qb = Qm + base;
  const bf16_t* Kb = Km + base;
  const bf16_t* Vb = Vm + base;
  float*        Ob = Om + base;

  const int q0    = wwin * 256 + wv * 32;  // first query row for this wave
  const int kwin0 = wwin * 256 - 256;      // first lookaround key position

  const int vkr = tid >> 3;        // 0..31 : key row this thread stages
  const int vd0 = (tid & 7) * 8;   // 0..56 : first head-dim column staged

  v8f   acc[2][4];
  float mstat[2][8], lstat[2][8];
#pragma unroll
  for (int mt = 0; mt < 2; ++mt)
#pragma unroll
    for (int r = 0; r < 8; ++r) {
      mstat[mt][r] = -3.0e38f;
      lstat[mt][r] = 0.f;
#pragma unroll
      for (int nt = 0; nt < 4; ++nt) acc[mt][nt][r] = 0.f;
    }

  for (int kc = 0; kc < 24; ++kc) {         // 768 keys in chunks of 32
    const int keyb = kc * 32;

    // --- stage V chunk transposed into LDS (vector load, scatter-store)
    __syncthreads();  // previous chunk's readers done
    {
      const int pos = clampi(kwin0 + keyb + vkr, 0, 2047);
      v8bf vrow = *(const v8bf*)(Vb + (size_t)pos * 1024 + vd0);
#pragma unroll
      for (int j = 0; j < 8; ++j) lV[vd0 + j][vkr] = vrow[j];
    }
    __syncthreads();

    // K fragments (B-matrix of QK^T): 2 key-subtiles x 2 k-steps over d=64
    v16bf kfr[2][2];
#pragma unroll
    for (int ns = 0; ns < 2; ++ns) {
      const int pos  = kwin0 + keyb + ns * 16 + l16;
      const int posc = clampi(pos, 0, 2047);
      const bf16_t* kp = Kb + (size_t)posc * 1024;
#pragma unroll
      for (int ks = 0; ks < 2; ++ks) {
        const bf16_t* q = kp + ks * 32 + h16 * 8;
        kfr[ns][ks] = pack16(*(const v8bf*)q, *(const v8bf*)(q + 16));
      }
    }

    // V fragments (B-matrix of P*V) from transposed LDS: contiguous reads
    v16bf vfr[4];
#pragma unroll
    for (int nt = 0; nt < 4; ++nt) {
      const bf16_t* vp = &lV[nt * 16 + l16][h16 * 8];
      vfr[nt] = pack16(*(const v8bf*)vp, *(const v8bf*)(vp + 16));
    }

#pragma unroll
    for (int mt = 0; mt < 2; ++mt) {
      // Q A-fragments
      const int qrow = q0 + mt * 16 + l16;
      const bf16_t* qp = Qb + (size_t)qrow * 1024;
      v16bf qfr[2];
#pragma unroll
      for (int ks = 0; ks < 2; ++ks) {
        const bf16_t* a = qp + ks * 32 + h16 * 8;
        qfr[ks] = pack16(*(const v8bf*)a, *(const v8bf*)(a + 16));
      }

      // sim = (Q * d^-0.5) K^T, masked at sequence boundaries
      float s[2][8];
#pragma unroll
      for (int ns = 0; ns < 2; ++ns) {
        v8f c;
#pragma unroll
        for (int r = 0; r < 8; ++r) c[r] = 0.f;
        c = WMMA_BF16(qfr[0], kfr[ns][0], c);
        c = WMMA_BF16(qfr[1], kfr[ns][1], c);
        const int kpos = kwin0 + keyb + ns * 16 + l16;   // lane's key column
        const bool valid = (unsigned)kpos < 2048u;
#pragma unroll
        for (int r = 0; r < 8; ++r)
          s[ns][r] = valid ? c[r] * 0.125f : -3.4e38f;
      }

      __builtin_amdgcn_wave_barrier();
      // online softmax over this 32-key chunk; row = r + 8*h16
#pragma unroll
      for (int r = 0; r < 8; ++r) {
        float mx = fmaxf(s[0][r], s[1][r]);
#pragma unroll
        for (int off = 8; off; off >>= 1) mx = fmaxf(mx, __shfl_xor(mx, off, 32));
        const float mo = mstat[mt][r];
        const float mn = fmaxf(mo, mx);
        const float corr = __expf(mo - mn);
        const float p0 = __expf(s[0][r] - mn);
        const float p1 = __expf(s[1][r] - mn);
        float rs = p0 + p1;
#pragma unroll
        for (int off = 8; off; off >>= 1) rs += __shfl_xor(rs, off, 32);
        mstat[mt][r] = mn;
        lstat[mt][r] = lstat[mt][r] * corr + rs;
#pragma unroll
        for (int nt = 0; nt < 4; ++nt) acc[mt][nt][r] *= corr;
        lP[wv][8 * h16 + r][l16]      = (bf16_t)p0;
        lP[wv][8 * h16 + r][16 + l16] = (bf16_t)p1;
      }
      __builtin_amdgcn_wave_barrier();

      // Re-read P in A-fragment order: row l16, ks = h16*8 + [0..7], +16..23
      const bf16_t* pp = &lP[wv][l16][h16 * 8];
      v16bf pfr = pack16(*(const v8bf*)pp, *(const v8bf*)(pp + 16));
      __builtin_amdgcn_wave_barrier();

#pragma unroll
      for (int nt = 0; nt < 4; ++nt)
        acc[mt][nt] = WMMA_BF16(pfr, vfr[nt], acc[mt][nt]);
    }
  }

  // epilogue: normalize and store (b, n, h*64 + d) fp32
#pragma unroll
  for (int mt = 0; mt < 2; ++mt)
#pragma unroll
    for (int r = 0; r < 8; ++r) {
      const float inv = 1.0f / lstat[mt][r];
      const int row = q0 + mt * 16 + h16 * 8 + r;
#pragma unroll
      for (int nt = 0; nt < 4; ++nt)
        Ob[(size_t)row * 1024 + nt * 16 + l16] = acc[mt][nt][r] * inv;
    }
}

// ---------------------------------------------------------------------------
// Rotary embedding (lucidrains variant, freqs = [f, f]) fused with f32->bf16.
// Input layout (b, n, h, d): d = i & 63, n = (i >> 10) & 2047.
// ---------------------------------------------------------------------------
__global__ __launch_bounds__(256) void rotary_cvt(
    const float* __restrict__ in, bf16_t* __restrict__ out, int rotate, int total)
{
  const int i = blockIdx.x * 256 + threadIdx.x;
  if (i >= total) return;
  float v = in[i];
  if (rotate) {
    const int d = i & 63;
    const int n = (i >> 10) & 2047;
    const int j = d & 31;
    const float inv = __expf(-9.210340372f * (float)j * (1.0f / 32.0f)); // 10000^(-j/32)
    const float ang = (float)n * inv;
    float sn, cs;
    __sincosf(ang, &sn, &cs);
    const float other = (d < 32) ? -in[i + 32] : in[i - 32];
    v = v * cs + other * sn;
  }
  out[i] = (bf16_t)v;
}

// ---------------------------------------------------------------------------
// out = LayerNorm(X + Y) * g + b ; one 256-thread block per 1024-wide row.
// ---------------------------------------------------------------------------
__global__ __launch_bounds__(256) void add_layernorm(
    const float* __restrict__ X, const float* __restrict__ Y,
    const float* __restrict__ g, const float* __restrict__ be,
    float* __restrict__ Of, bf16_t* __restrict__ Ob)
{
  __shared__ float red1[8], red2[8];
  const int row = blockIdx.x;
  const int tid = threadIdx.x;
  const float* x = X + (size_t)row * 1024;
  const float* y = Y + (size_t)row * 1024;

  float a[4];
  float s = 0.f;
#pragma unroll
  for (int i = 0; i < 4; ++i) { a[i] = x[tid + i * 256] + y[tid + i * 256]; s += a[i]; }
#pragma unroll
  for (int off = 16; off; off >>= 1) s += __shfl_xor(s, off, 32);
  if ((tid & 31) == 0) red1[tid >> 5] = s;
  __syncthreads();
  float mu = 0.f;
#pragma unroll
  for (int w = 0; w < 8; ++w) mu += red1[w];
  mu *= (1.0f / 1024.0f);

  float vs = 0.f;
#pragma unroll
  for (int i = 0; i < 4; ++i) { const float d = a[i] - mu; vs += d * d; }
#pragma unroll
  for (int off = 16; off; off >>= 1) vs += __shfl_xor(vs, off, 32);
  if ((tid & 31) == 0) red2[tid >> 5] = vs;
  __syncthreads();
  float var = 0.f;
#pragma unroll
  for (int w = 0; w < 8; ++w) var += red2[w];
  var *= (1.0f / 1024.0f);
  const float rs = rsqrtf(var + 1e-5f);

#pragma unroll
  for (int i = 0; i < 4; ++i) {
    const int c = tid + i * 256;
    const float o = (a[i] - mu) * rs * g[c] + be[c];
    Of[(size_t)row * 1024 + c] = o;
    if (Ob) Ob[(size_t)row * 1024 + c] = (bf16_t)o;
  }
}

__global__ __launch_bounds__(256) void cvt_bf16(
    const float* __restrict__ in, bf16_t* __restrict__ out, int n)
{
  const int i = blockIdx.x * 256 + threadIdx.x;
  if (i < n) out[i] = (bf16_t)in[i];
}

// ---------------------------------------------------------------------------
// Host orchestration. Workspace usage ~160 MB.
// Input order: src, then per layer (x2):
//   wq,bq,wk,bk,wv,bv,wo,bo,w1,b1,w2,b2,g1,be1,g2,be2
// ---------------------------------------------------------------------------
extern "C" void kernel_launch(void* const* d_in, const int* in_sizes, int n_in,
                              void* d_out, int out_size, void* d_ws, size_t ws_size,
                              hipStream_t stream)
{
  (void)in_sizes; (void)n_in; (void)out_size; (void)ws_size;
  const size_t R   = 4096;           // B*N rows
  const size_t DM  = 1024;
  const size_t DFF = 4096;
  const int    TOT = (int)(R * DM);  // 4,194,304

  const float* src = (const float*)d_in[0];

  char* p = (char*)d_ws;
  auto carve = [&](size_t bytes) -> void* {
    void* r = (void*)p;
    p += (bytes + 255) & ~(size_t)255;
    return r;
  };

  bf16_t* Wb  = (bf16_t*)carve((size_t)12582912 * 2);  // per-layer bf16 weights
  float*  X   = (float*) carve(R * DM * 4);
  bf16_t* Xb  = (bf16_t*)carve(R * DM * 2);
  float*  S1  = (float*) carve(R * DM * 4);
  float*  S2  = (float*) carve(R * DM * 4);
  float*  S3  = (float*) carve(R * DM * 4);
  bf16_t* Qbf = (bf16_t*)carve(R * DM * 2);
  bf16_t* Kbf = (bf16_t*)carve(R * DM * 2);
  bf16_t* Vbf = (bf16_t*)carve(R * DM * 2);
  bf16_t* Hb  = (bf16_t*)carve(R * DFF * 2);

  hipMemcpyAsync(X, src, R * DM * 4, hipMemcpyDeviceToDevice, stream);

  const dim3 blk(256);
  auto cgrid = [](int n) { return dim3((unsigned)((n + 255) / 256)); };
  const dim3 gN1024(32, 16), gN4096(32, 64);

  for (int li = 0; li < 2; ++li) {
    const int ib = 1 + li * 16;
    const float* wq  = (const float*)d_in[ib + 0];
    const float* bq  = (const float*)d_in[ib + 1];
    const float* wk  = (const float*)d_in[ib + 2];
    const float* bk  = (const float*)d_in[ib + 3];
    const float* wvp = (const float*)d_in[ib + 4];
    const float* bv  = (const float*)d_in[ib + 5];
    const float* wo  = (const float*)d_in[ib + 6];
    const float* bo  = (const float*)d_in[ib + 7];
    const float* w1  = (const float*)d_in[ib + 8];
    const float* b1  = (const float*)d_in[ib + 9];
    const float* w2  = (const float*)d_in[ib + 10];
    const float* b2  = (const float*)d_in[ib + 11];
    const float* g1  = (const float*)d_in[ib + 12];
    const float* be1 = (const float*)d_in[ib + 13];
    const float* g2  = (const float*)d_in[ib + 14];
    const float* be2 = (const float*)d_in[ib + 15];

    bf16_t* wqb = Wb + 0;
    bf16_t* wkb = Wb + 1048576;
    bf16_t* wvb = Wb + 2097152;
    bf16_t* wob = Wb + 3145728;
    bf16_t* w1b = Wb + 4194304;
    bf16_t* w2b = Wb + 8388608;

    cvt_bf16<<<cgrid(1048576), blk, 0, stream>>>(wq,  wqb, 1048576);
    cvt_bf16<<<cgrid(1048576), blk, 0, stream>>>(wk,  wkb, 1048576);
    cvt_bf16<<<cgrid(1048576), blk, 0, stream>>>(wvp, wvb, 1048576);
    cvt_bf16<<<cgrid(1048576), blk, 0, stream>>>(wo,  wob, 1048576);
    cvt_bf16<<<cgrid(4194304), blk, 0, stream>>>(w1,  w1b, 4194304);
    cvt_bf16<<<cgrid(4194304), blk, 0, stream>>>(w2,  w2b, 4194304);
    cvt_bf16<<<cgrid(TOT), blk, 0, stream>>>(X, Xb, TOT);

    // Q,K,V projections (fp32 out for rotary precision)
    gemm_nt<<<gN1024, blk, 0, stream>>>(Xb, wqb, bq, S1, nullptr, 4096, 1024, 1024, 0);
    gemm_nt<<<gN1024, blk, 0, stream>>>(Xb, wkb, bk, S2, nullptr, 4096, 1024, 1024, 0);
    gemm_nt<<<gN1024, blk, 0, stream>>>(Xb, wvb, bv, S3, nullptr, 4096, 1024, 1024, 0);
    rotary_cvt<<<cgrid(TOT), blk, 0, stream>>>(S1, Qbf, 1, TOT);
    rotary_cvt<<<cgrid(TOT), blk, 0, stream>>>(S2, Kbf, 1, TOT);
    rotary_cvt<<<cgrid(TOT), blk, 0, stream>>>(S3, Vbf, 0, TOT);

    // windowed attention -> S1 (fp32, heads concatenated)
    local_attn<<<dim3(256), blk, 0, stream>>>(Qbf, Kbf, Vbf, S1);

    // output projection + residual LN
    cvt_bf16<<<cgrid(TOT), blk, 0, stream>>>(S1, Xb, TOT);
    gemm_nt<<<gN1024, blk, 0, stream>>>(Xb, wob, bo, S2, nullptr, 4096, 1024, 1024, 0);
    add_layernorm<<<dim3(4096), blk, 0, stream>>>(X, S2, g1, be1, S3, Qbf); // X2 f32=S3, bf16=Qbf

    // FFN: relu(X2 W1^T + b1) W2^T + b2, then residual LN
    gemm_nt<<<gN4096, blk, 0, stream>>>(Qbf, w1b, b1, nullptr, Hb, 4096, 4096, 1024, 1);
    gemm_nt<<<gN1024, blk, 0, stream>>>(Hb,  w2b, b2, S1, nullptr, 4096, 1024, 4096, 0);

    float* Xn = (li == 1) ? (float*)d_out : X;
    add_layernorm<<<dim3(4096), blk, 0, stream>>>(S3, S1, g2, be2, Xn, nullptr);
  }
}
